// Method_Generation_69200513073790
// MI455X (gfx1250) — compile-verified, matrix operand order checked by reading, not run on hardware
//
#include <hip/hip_runtime.h>

typedef __attribute__((ext_vector_type(16))) _Float16 v16h;
typedef __attribute__((ext_vector_type(8)))  _Float16 v8h;
typedef __attribute__((ext_vector_type(8)))  float    v8f;
typedef __attribute__((ext_vector_type(4)))  float    f32x4;

#define Bq   5000
#define Tq   8
#define Vq   5264
#define Hq   200
#define BTq  40000      // B*T
#define KP1  5280       // V padded to 165*32
#define NKB  165        // K blocks in phase 1
#define HP   208        // H padded rows (13*16)
#define HKP  224        // H padded as K dim (7*32)
#define NT_H 13         // N tiles covering 208
#define MT3  313        // ceil(5000/16)
#define NT3  329        // 5264/16
#define AROW 36         // f32 LDS tile row stride (144B: b128-aligned, conflict-free)

__device__ inline v8f zero8() {
    v8f z = {0.f, 0.f, 0.f, 0.f, 0.f, 0.f, 0.f, 0.f};
    return z;
}

__device__ inline v16h cat8(v8h lo, v8h hi) {
    return __builtin_shufflevector(lo, hi, 0,1,2,3,4,5,6,7,8,9,10,11,12,13,14,15);
}

// B fragment (32x16 f16) direct from a padded f16 weight matrix, row-major
// [N rows][ldk]. Lane<16: col n=lane, K{kb..kb+7, kb+16..kb+23}; lane>=16 shifts K by 8.
__device__ inline v16h ldB_frag(const _Float16* w, int ldk, int nrow, int kb, int lane) {
    const _Float16* p = w + (size_t)nrow * ldk + kb + ((lane & 16) ? 8 : 0);
    v8h lo = *(const v8h*)(p);
    v8h hi = *(const v8h*)(p + 16);
    return cat8(lo, hi);
}

// A fragment from an fp32 matrix (row stride ld), converting to f16 in-register.
// K chunks of 8 are all-or-nothing valid since kvalid (200) is a multiple of 8.
__device__ inline v16h ldA_f32_frag(const float* a, int ld, int m, int kb, int kvalid, int lane) {
    int s0 = kb + ((lane & 16) ? 8 : 0);
    int s1 = s0 + 16;
    const float* row = a + (size_t)m * ld;
    v16h r;
    if (s0 < kvalid) {
        f32x4 x0 = *(const f32x4*)(row + s0);
        f32x4 x1 = *(const f32x4*)(row + s0 + 4);
        r[0]=(_Float16)x0[0]; r[1]=(_Float16)x0[1]; r[2]=(_Float16)x0[2]; r[3]=(_Float16)x0[3];
        r[4]=(_Float16)x1[0]; r[5]=(_Float16)x1[1]; r[6]=(_Float16)x1[2]; r[7]=(_Float16)x1[3];
    } else {
        r[0]=r[1]=r[2]=r[3]=r[4]=r[5]=r[6]=r[7]=(_Float16)0.f;
    }
    if (s1 < kvalid) {
        f32x4 x0 = *(const f32x4*)(row + s1);
        f32x4 x1 = *(const f32x4*)(row + s1 + 4);
        r[8]=(_Float16)x0[0];  r[9]=(_Float16)x0[1];  r[10]=(_Float16)x0[2]; r[11]=(_Float16)x0[3];
        r[12]=(_Float16)x1[0]; r[13]=(_Float16)x1[1]; r[14]=(_Float16)x1[2]; r[15]=(_Float16)x1[3];
    } else {
        r[8]=r[9]=r[10]=r[11]=r[12]=r[13]=r[14]=r[15]=(_Float16)0.f;
    }
    return r;
}

// ---------------- weight conversion kernels (fp32 -> padded f16) ----------------

__global__ void cvt_wih_kernel(const float* __restrict__ W, _Float16* __restrict__ w16) {
    int idx = blockIdx.x * blockDim.x + threadIdx.x;
    if (idx >= HP * KP1) return;
    int n = idx / KP1, k = idx - n * KP1;
    float v = (n < Hq && k < Vq) ? W[(size_t)n * Vq + k] : 0.f;
    w16[idx] = (_Float16)v;
}

__global__ void cvt_wfc_kernel(const float* __restrict__ W, _Float16* __restrict__ w16) {
    int idx = blockIdx.x * blockDim.x + threadIdx.x;
    if (idx >= Vq * HKP) return;
    int n = idx / HKP, k = idx - n * HKP;
    float v = (k < Hq) ? W[(size_t)n * Hq + k] : 0.f;
    w16[idx] = (_Float16)v;
}

__global__ void cvt_whh_kernel(const float* __restrict__ W, _Float16* __restrict__ w16) {
    int idx = blockIdx.x * blockDim.x + threadIdx.x;
    if (idx >= HP * HKP) return;
    int n = idx / HKP, k = idx - n * HKP;
    float v = (n < Hq && k < Hq) ? W[(size_t)n * Hq + k] : 0.f;
    w16[idx] = (_Float16)v;
}

// ---------------- phase 1: xh[t][b][h] = X·W_ih^T + b_ih + b_hh ----------------
// One wave per 16-row M tile. X is streamed HBM->LDS with double-buffered
// GLOBAL_LOAD_ASYNC_TO_LDS_B128 (ASYNCcnt), f32->f16 conversion happens after the
// ds_load fragment gather. 13 fp32 accumulators cover N=200, so X is read once.

__global__ void __launch_bounds__(128) gemm1_kernel(const float* __restrict__ X,
                                                    const _Float16* __restrict__ w16,
                                                    const float* __restrict__ b_ih,
                                                    const float* __restrict__ b_hh,
                                                    float* __restrict__ xh) {
    __shared__ __align__(16) float sA[4][2][16 * AROW];
    int lane = threadIdx.x & 31;
    int wv   = threadIdx.x >> 5;
    int mtile = blockIdx.x * 4 + wv;            // 625*4 == 2500 exactly
    long m0 = (long)mtile * 16;

    v8f acc[NT_H];
#pragma unroll
    for (int i = 0; i < NT_H; ++i) acc[i] = zero8();

    int srow = lane >> 3;          // 0..3
    int scol = (lane & 7) * 4;     // 0,4,...,28

    // issue one 16x32 fp32 tile as 4 async b128 transfers (512B each, whole wave)
    auto issue_tile = [&](float* buf, int kb) {
#pragma unroll
        for (int i = 0; i < 4; ++i) {
            int row = i * 4 + srow;
            int k = kb + scol;
            if (k >= Vq) k = 0;    // clamp: pairs with zero-padded B rows -> contributes 0
            const float* g = X + (m0 + row) * (long)Vq + k;
            unsigned loff = (unsigned)(unsigned long long)(buf + row * AROW + scol);
            asm volatile("global_load_async_to_lds_b128 %0, %1, off"
                         :: "v"(loff), "v"(g) : "memory");
        }
    };

    issue_tile(&sA[wv][0][0], 0);
    int row = lane & 15;
    int k0  = (lane & 16) ? 8 : 0;
    int ncl = lane & 15;

    for (int ib = 0; ib < NKB; ++ib) {
        int nb = (ib + 1 < NKB) ? ib + 1 : 0;             // wrap: dummy prefetch, never consumed
        issue_tile(&sA[wv][(ib + 1) & 1][0], nb * 32);
        asm volatile("s_wait_asynccnt 0x4" ::: "memory"); // oldest 4 (current tile) complete

        const float* tile = &sA[wv][ib & 1][0];
        f32x4 a0 = *(const f32x4*)(tile + row * AROW + k0);
        f32x4 a1 = *(const f32x4*)(tile + row * AROW + k0 + 4);
        f32x4 a2 = *(const f32x4*)(tile + row * AROW + k0 + 16);
        f32x4 a3 = *(const f32x4*)(tile + row * AROW + k0 + 20);
        v16h a;
        a[0]=(_Float16)a0[0];  a[1]=(_Float16)a0[1];  a[2]=(_Float16)a0[2];  a[3]=(_Float16)a0[3];
        a[4]=(_Float16)a1[0];  a[5]=(_Float16)a1[1];  a[6]=(_Float16)a1[2];  a[7]=(_Float16)a1[3];
        a[8]=(_Float16)a2[0];  a[9]=(_Float16)a2[1];  a[10]=(_Float16)a2[2]; a[11]=(_Float16)a2[3];
        a[12]=(_Float16)a3[0]; a[13]=(_Float16)a3[1]; a[14]=(_Float16)a3[2]; a[15]=(_Float16)a3[3];

        int kb = ib * 32;
#pragma unroll
        for (int nt = 0; nt < NT_H; ++nt) {
            v16h b = ldB_frag(w16, KP1, nt * 16 + ncl, kb, lane);
            acc[nt] = __builtin_amdgcn_wmma_f32_16x16x32_f16(
                false, a, false, b, (short)0, acc[nt], false, false);
        }
    }

    int rofs = (lane & 16) ? 8 : 0;
#pragma unroll
    for (int nt = 0; nt < NT_H; ++nt) {
        int n = nt * 16 + ncl;
        if (n < Hq) {
            float bias = b_ih[n] + b_hh[n];
#pragma unroll
            for (int v = 0; v < 8; ++v) {
                long r = m0 + v + rofs;          // r = b*T + t
                long b_ = r >> 3;
                long t_ = r & 7;
                xh[(t_ * Bq + b_) * Hq + n] = acc[nt][v] + bias;   // [t][b][h]
            }
        }
    }
}

// ---------------- phase 2: h = tanh(xh_t + h·W_hh^T), one step ----------------
// h kept fp32 in memory; A fragment converted in-register. In-place safe: each
// wave reads all 16 of its rows into registers before any store.

__global__ void __launch_bounds__(128) rnn_step_kernel(const float* h_in,
                                                       float* h_out,
                                                       const float* __restrict__ xh_t,
                                                       const _Float16* __restrict__ whh16) {
    int lane = threadIdx.x & 31;
    int wv   = threadIdx.x >> 5;
    int mtile = blockIdx.x * 4 + wv;
    if (mtile >= MT3) return;
    int m0 = mtile * 16;
    int mload = m0 + (lane & 15);
    if (mload >= Bq) mload = Bq - 1;

    v16h afrag[7];
#pragma unroll
    for (int kb = 0; kb < 7; ++kb)
        afrag[kb] = ldA_f32_frag(h_in, Hq, mload, kb * 32, Hq, lane);

    int ncol = lane & 15;
    int rofs = (lane & 16) ? 8 : 0;
#pragma unroll
    for (int nt = 0; nt < NT_H; ++nt) {
        v8f acc = zero8();
#pragma unroll
        for (int kb = 0; kb < 7; ++kb) {
            v16h b = ldB_frag(whh16, HKP, nt * 16 + ncol, kb * 32, lane);
            acc = __builtin_amdgcn_wmma_f32_16x16x32_f16(
                false, afrag[kb], false, b, (short)0, acc, false, false);
        }
        int n = nt * 16 + ncol;
        if (n < Hq) {
#pragma unroll
            for (int v = 0; v < 8; ++v) {
                int r = m0 + v + rofs;
                if (r < Bq) {
                    float pre = acc[v] + xh_t[(long)r * Hq + n];   // contiguous slab
                    h_out[(long)r * Hq + n] = tanhf(pre);
                }
            }
        }
    }
}

// ---------------- phase 3: out = h_last·W_fc^T + b_fc ----------------
// A fragment (16x200) held in registers for all N tiles; W_fc16 is L2-resident.

__global__ void __launch_bounds__(128) gemm3_kernel(const float* __restrict__ h_last,
                                                    const _Float16* __restrict__ wfc16,
                                                    const float* __restrict__ b_fc,
                                                    float* __restrict__ out) {
    int lane = threadIdx.x & 31;
    int wv   = threadIdx.x >> 5;
    int mtile = blockIdx.x * 4 + wv;
    if (mtile >= MT3) return;
    int m0 = mtile * 16;
    int mload = m0 + (lane & 15);
    if (mload >= Bq) mload = Bq - 1;

    v16h afrag[7];
#pragma unroll
    for (int kb = 0; kb < 7; ++kb)
        afrag[kb] = ldA_f32_frag(h_last, Hq, mload, kb * 32, Hq, lane);

    int ntBeg = blockIdx.y * 42;
    int ntEnd = ntBeg + 42;
    if (ntEnd > NT3) ntEnd = NT3;

    int ncol = lane & 15;
    int rofs = (lane & 16) ? 8 : 0;
    for (int nt = ntBeg; nt < ntEnd; ++nt) {
        v8f acc = zero8();
#pragma unroll
        for (int kb = 0; kb < 7; ++kb) {
            v16h b = ldB_frag(wfc16, HKP, nt * 16 + ncol, kb * 32, lane);
            acc = __builtin_amdgcn_wmma_f32_16x16x32_f16(
                false, afrag[kb], false, b, (short)0, acc, false, false);
        }
        int n = nt * 16 + ncol;
        float bias = b_fc[n];
#pragma unroll
        for (int v = 0; v < 8; ++v) {
            int r = m0 + v + rofs;
            if (r < Bq) out[(long)r * Vq + n] = acc[v] + bias;
        }
    }
}

// ---------------- host launcher ----------------

extern "C" void kernel_launch(void* const* d_in, const int* in_sizes, int n_in,
                              void* d_out, int out_size, void* d_ws, size_t ws_size,
                              hipStream_t stream) {
    (void)in_sizes; (void)n_in; (void)out_size; (void)ws_size;
    const float* X    = (const float*)d_in[0];
    const float* h0   = (const float*)d_in[1];
    const float* W_ih = (const float*)d_in[2];
    const float* W_hh = (const float*)d_in[3];
    const float* b_ih = (const float*)d_in[4];
    const float* b_hh = (const float*)d_in[5];
    const float* W_fc = (const float*)d_in[6];
    const float* b_fc = (const float*)d_in[7];
    float* out = (float*)d_out;

    char* ws = (char*)d_ws;
    // 256B-aligned workspace carve-up (~40.6 MB total)
    _Float16* wih16 = (_Float16*)(ws + 0);            // 208*5280*2 = 2,196,480
    _Float16* wfc16 = (_Float16*)(ws + 2196480);      // 5264*224*2 = 2,358,272
    _Float16* whh16 = (_Float16*)(ws + 4554752);      //  208*224*2 =    93,184
    float*    xh    = (float*)   (ws + 4647936);      // 8*5000*200*4 = 32,000,000  [t][b][h]
    float*    h     = (float*)   (ws + 36647936);     //  5000*200*4 =  4,000,000

    cvt_wih_kernel<<<(HP * KP1 + 255) / 256, 256, 0, stream>>>(W_ih, wih16);
    cvt_wfc_kernel<<<(Vq * HKP + 255) / 256, 256, 0, stream>>>(W_fc, wfc16);
    cvt_whh_kernel<<<(HP * HKP + 255) / 256, 256, 0, stream>>>(W_hh, whh16);

    gemm1_kernel<<<BTq / (16 * 4), 128, 0, stream>>>(X, wih16, b_ih, b_hh, xh);

    for (int t = 0; t < Tq; ++t) {
        const float* hin = (t == 0) ? h0 : (const float*)h;
        const float* xh_t = xh + (size_t)t * Bq * Hq;
        rnn_step_kernel<<<(MT3 + 3) / 4, 128, 0, stream>>>(hin, h, xh_t, whh16);
    }

    gemm3_kernel<<<dim3((MT3 + 3) / 4, 8), 128, 0, stream>>>(h, wfc16, b_fc, out);
}
